// RewardPredictorEnsemble_20822001451489
// MI455X (gfx1250) — matrix operand order repaired and checked by reading, not compiled, not run on hardware
//
#include <hip/hip_runtime.h>

typedef __attribute__((ext_vector_type(16))) _Float16 v16h;
typedef __attribute__((ext_vector_type(8)))  _Float16 v8h;
typedef __attribute__((ext_vector_type(4)))  _Float16 v4h;
typedef __attribute__((ext_vector_type(2)))  _Float16 v2h;
typedef __attribute__((ext_vector_type(8)))  float    v8f;

#define AST 72   // row stride in halves: 144B rows -> 16B aligned, bank-conflict free

// wave-local LDS ordering fence: DS ops are in-order per wave; the "memory"
// clobber stops the compiler reordering cross-lane store->load sequences.
#define WAVE_LDS_FENCE() asm volatile("s_wait_dscnt 0" ::: "memory")

// ---- WMMA operand builders (ISA 7.12.2 16-bit layouts), all from LDS ----

// A (16x32 f16): lane m(0..15)=row m; lanes<16: K=k0..k0+7 then k0+16..23; lanes>=16: +8
__device__ __forceinline__ v16h load_A(const _Float16* base, int lane, int k0) {
  const int r = lane & 15, hi = lane >> 4;
  const _Float16* p0 = base + r * AST + k0 + hi * 8;
  v8h c0 = *(const v8h*)(p0);        // ds_load_b128
  v8h c1 = *(const v8h*)(p0 + 16);   // ds_load_b128
  v16h a;
#pragma unroll
  for (int i = 0; i < 8; ++i) { a[i] = c0[i]; a[i + 8] = c1[i]; }
  return a;
}

// B (32x16 f16) from transposed weights WT[col][k]: lane n=col n; lanes<16: K=k0..k0+7 then k0+16..23; lanes>=16: +8
__device__ __forceinline__ v16h load_B(const _Float16* wt, int lane, int ct, int k0) {
  const int n = lane & 15, hi = lane >> 4;
  const _Float16* p0 = wt + (ct * 16 + n) * AST + k0 + hi * 8;
  v8h c0 = *(const v8h*)(p0);        // ds_load_b128
  v8h c1 = *(const v8h*)(p0 + 16);   // ds_load_b128
  v16h b;
#pragma unroll
  for (int i = 0; i < 8; ++i) { b[i] = c0[i]; b[i + 8] = c1[i]; }
  return b;
}

// One dense layer for a 16-row tile: out[16xN] = leaky(in[16xK] @ W[KxN] + bias)
// Phase 1: all column-tile WMMA chains back-to-back (no D->A/B hazards between
// them, XDL stays busy). Phase 2: packed-f16 epilogues (tile i's epilogue fills
// the WMMA->VALU hazard window of tile i+1's chain).
template<int K, int N>
__device__ __forceinline__ void run_layer(const _Float16* in, _Float16* out,
                                          const _Float16* wt, const float* bias, int lane) {
  constexpr int NCT = N / 16;
  const int col = lane & 15, hi = lane >> 4;
  v16h a0 = load_A(in, lane, 0);
  v16h a1{};
  if constexpr (K == 64) a1 = load_A(in, lane, 32);

  v8f accs[NCT];
#pragma unroll
  for (int ct = 0; ct < NCT; ++ct) {
    v8f acc = {};
    v16h b0 = load_B(wt, lane, ct, 0);
    acc = __builtin_amdgcn_wmma_f32_16x16x32_f16(false, a0, false, b0, (short)0, acc, false, false);
    if constexpr (K == 64) {
      v16h b1 = load_B(wt, lane, ct, 32);
      acc = __builtin_amdgcn_wmma_f32_16x16x32_f16(false, a1, false, b1, (short)0, acc, false, false);
    }
    accs[ct] = acc;
  }

  const v2h slope2 = {(_Float16)0.01f, (_Float16)0.01f};
#pragma unroll
  for (int ct = 0; ct < NCT; ++ct) {
    const _Float16 bvh = (_Float16)bias[ct * 16 + col];
    const v2h bv2 = {bvh, bvh};
    // D layout: VGPR v -> (row v, col lane) lanes 0-15 ; (row v+8, col lane-16) lanes 16-31
#pragma unroll
    for (int vp = 0; vp < 4; ++vp) {
      v2h pr = {(_Float16)accs[ct][2 * vp], (_Float16)accs[ct][2 * vp + 1]};  // v_cvt_pk_f16_f32
      pr = pr + bv2;                                                 // v_pk_add_f16
      v2h lk = pr * slope2;                                          // v_pk_mul_f16
      pr = __builtin_elementwise_max(pr, lk);                        // v_pk_max_num_f16
      out[(2 * vp     + hi * 8) * AST + ct * 16 + col] = pr[0];      // ds_store_b16
      out[(2 * vp + 1 + hi * 8) * AST + ct * 16 + col] = pr[1];      // ds_store_b16_d16_hi
    }
  }
}

// stage one weight matrix W[K][N] (f32, row-major) -> LDS WT[n][k] (f16), K-pairs packed
__device__ __forceinline__ void stage_wt(const float* __restrict__ Wg, _Float16* dst,
                                         int K, int N, int tid) {
  const int pairs = (K * N) >> 1;
  for (int i = tid; i < pairs; i += 256) {
    int kp = i / N, n = i - kp * N;
    float a = Wg[(2 * kp) * N + n];          // coalesced across threads (consecutive n)
    float b = Wg[(2 * kp + 1) * N + n];
    v2h pr = {(_Float16)a, (_Float16)b};
    *(v2h*)(dst + n * AST + 2 * kp) = pr;    // one b32 store for two halves
  }
}

__global__ __launch_bounds__(256)
void ens_reward_kernel(const float* __restrict__ s1, const float* __restrict__ s2,
                       const float* __restrict__ W1, const float* __restrict__ b1,
                       const float* __restrict__ W2, const float* __restrict__ b2,
                       const float* __restrict__ W3, const float* __restrict__ b3,
                       const float* __restrict__ W4, const float* __restrict__ b4,
                       const float* __restrict__ W5, const float* __restrict__ b5,
                       float* __restrict__ out) {
  // LDS: 36864 + 25344 + 772 + 64 = 63044 B
  __shared__ __align__(16) _Float16 s_act[8][2][16 * AST];   // per-wave ping/pong activation tiles
  __shared__ __align__(16) _Float16 s_wt[176 * AST];         // transposed f16 weights (W1..W4)
  __shared__ __align__(16) float    s_bias[193];             // b1@0 b2@64 b3@128 b4@160 w5@176 b5@192
  __shared__ float s_red[8][2];

  constexpr int W1OFF = 0, W2OFF = 64 * AST, W3OFF = 128 * AST, W4OFF = 160 * AST;

  const int tid  = threadIdx.x;
  const int lane = tid & 31;
  const int w    = tid >> 5;                 // wave id 0..7 -> frame tile
  const int bidx = blockIdx.x;               // batch; WG covers both s tensors
  const size_t rowoff = (size_t)(bidx * 128 + w * 16) * 64;
  const float* xs[2] = { s1 + rowoff, s2 + rowoff };

  _Float16* ping = &s_act[w][0][0];
  _Float16* pong = &s_act[w][1][0];

  float acc2[2] = {0.0f, 0.0f};              // valid on lane 0 only

  for (int p = 0; p < 8; ++p) {
    __syncthreads();
    // ---- cooperative stage of predictor p weights (f16, transposed, pair-packed) ----
    stage_wt(W1 + p * 4096, s_wt + W1OFF, 64, 64, tid);
    stage_wt(W2 + p * 4096, s_wt + W2OFF, 64, 64, tid);
    stage_wt(W3 + p * 2048, s_wt + W3OFF, 64, 32, tid);
    stage_wt(W4 + p * 512,  s_wt + W4OFF, 32, 16, tid);
    if (tid < 64) s_bias[tid]       = b1[p * 64 + tid];
    if (tid < 64) s_bias[64 + tid]  = b2[p * 64 + tid];
    if (tid < 32) s_bias[128 + tid] = b3[p * 32 + tid];
    if (tid < 16) s_bias[160 + tid] = b4[p * 16 + tid];
    if (tid < 16) s_bias[176 + tid] = W5[p * 16 + tid];     // final weights kept in f32
    if (tid == 0) s_bias[192]       = b5[p];
    __syncthreads();

    // ---- run both tensors through predictor p (weights staged once) ----
#pragma unroll
    for (int s = 0; s < 2; ++s) {
      const float* xr = xs[s];
      // convert this wave's 16x64 input tile to f16: float4 loads -> v4h stores
#pragma unroll
      for (int i = lane; i < 256; i += 32) {
        int r = i >> 4, cq = i & 15;
        const float4 v = *(const float4*)(xr + r * 64 + cq * 4);   // global_load_b128
        v4h h = {(_Float16)v.x, (_Float16)v.y, (_Float16)v.z, (_Float16)v.w};
        *(v4h*)(ping + r * AST + cq * 4) = h;                      // ds_store_b64
      }
      WAVE_LDS_FENCE();
      run_layer<64, 64>(ping, pong, s_wt + W1OFF, s_bias + 0,   lane);
      WAVE_LDS_FENCE();
      run_layer<64, 64>(pong, ping, s_wt + W2OFF, s_bias + 64,  lane);
      WAVE_LDS_FENCE();
      run_layer<64, 32>(ping, pong, s_wt + W3OFF, s_bias + 128, lane);
      WAVE_LDS_FENCE();

      // ---- layer 4 (32->16): keep D in registers ----
      v16h a0 = load_A(pong, lane, 0);                 // h3: 16x32
      v16h b0 = load_B(s_wt + W4OFF, lane, 0, 0);      // W4T: 16 cols x K=32
      v8f acc = {};
      acc = __builtin_amdgcn_wmma_f32_16x16x32_f16(false, a0, false, b0, (short)0, acc, false, false);

      // ---- layer 5 (16->1) fused: D is (sample v/v+8, feat lane&15); the output
      //      is summed over samples AND features anyway -> per-lane dot + wave reduce.
      const float b4v = s_bias[160 + (lane & 15)];
      const float w5v = s_bias[176 + (lane & 15)];
      float t = 0.0f;
#pragma unroll
      for (int v = 0; v < 8; ++v) {
        float x = acc[v] + b4v;
        x = fmaxf(x, 0.01f * x);                       // LeakyReLU(0.01)
        t += x * w5v;
      }
#pragma unroll
      for (int m = 1; m <= 16; m <<= 1)                // full wave32 reduction
        t += __shfl_xor(t, m, 32);
      if (lane == 0) acc2[s] += t + 16.0f * s_bias[192];   // + b5 per each of 16 samples
    }
  }

  // ---- reduce across the 8 waves (acc2 valid on lane 0) ----
  if (lane == 0) { s_red[w][0] = acc2[0]; s_red[w][1] = acc2[1]; }
  __syncthreads();
  if (tid < 2) {
    float s = 0.0f;
#pragma unroll
    for (int i = 0; i < 8; ++i) s += s_red[i][tid];
    out[bidx * 2 + tid] = s * 0.125f;                  // divide by P=8
  }
}

extern "C" void kernel_launch(void* const* d_in, const int* in_sizes, int n_in,
                              void* d_out, int out_size, void* d_ws, size_t ws_size,
                              hipStream_t stream) {
  const float* s1 = (const float*)d_in[0];
  const float* s2 = (const float*)d_in[1];
  const float* W1 = (const float*)d_in[2];
  const float* b1 = (const float*)d_in[3];
  const float* W2 = (const float*)d_in[4];
  const float* b2 = (const float*)d_in[5];
  const float* W3 = (const float*)d_in[6];
  const float* b3 = (const float*)d_in[7];
  const float* W4 = (const float*)d_in[8];
  const float* b4 = (const float*)d_in[9];
  const float* W5 = (const float*)d_in[10];
  const float* b5 = (const float*)d_in[11];
  float* out = (float*)d_out;

  const int B = in_sizes[0] / (128 * 64);        // 1024
  dim3 grid(B), block(256);
  ens_reward_kernel<<<grid, block, 0, stream>>>(s1, s2, W1, b1, W2, b2, W3, b3,
                                                W4, b4, W5, b5, out);
}
